// GATEncoder_62577673503714
// MI455X (gfx1250) — matrix operand order repaired and checked by reading, not compiled, not run on hardware
//
#include <hip/hip_runtime.h>

typedef __attribute__((ext_vector_type(16))) __bf16 v16bf;
typedef __attribute__((ext_vector_type(8)))  float  v8f;

#define NNODES 100000
#define NEDGES 400000
#define ETOT   (NEDGES + NNODES)   // edges + self loops
#define MTILES (NNODES / 16)       // 6250, exact

// ---------- helpers ----------
__device__ __forceinline__ unsigned f2ord(float f) {
  unsigned u = __float_as_uint(f);
  return (u & 0x80000000u) ? ~u : (u | 0x80000000u);   // order-preserving map
}
__device__ __forceinline__ float ord2f(unsigned u) {
  return __uint_as_float((u & 0x80000000u) ? (u & 0x7fffffffu) : ~u);
}

// ---------- utility kernels ----------
__global__ void k_zero_f32(float* __restrict__ p, long n) {
  long i = (long)blockIdx.x * blockDim.x + threadIdx.x;
  long st = (long)gridDim.x * blockDim.x;
  for (; i < n; i += st) p[i] = 0.0f;
}

__global__ void k_f32_to_bf16(const float* __restrict__ in, __bf16* __restrict__ out, long n) {
  long i = (long)blockIdx.x * blockDim.x + threadIdx.x;
  long st = (long)gridDim.x * blockDim.x;
  for (; i < n; i += st) out[i] = (__bf16)in[i];
}

// W [K, Nc] f32  ->  Wt [Nc, K] bf16  (so B fragments read contiguous K)
__global__ void k_transpose_bf16(const float* __restrict__ W, __bf16* __restrict__ Wt,
                                 int K, int Nc) {
  int i = blockIdx.x * blockDim.x + threadIdx.x;
  if (i >= K * Nc) return;
  int k = i / Nc, n = i % Nc;
  Wt[(long)n * K + k] = (__bf16)W[i];
}

// ---------- WMMA GEMM: C[M,Nc] = A_bf16[M,K] @ W_bf16[K,Nc]  (Wt is [Nc,K]) ----------
// Block = 256 threads = 8 waves, each wave owns a 16-row m-tile and 8 n-tiles (128 cols).
// The 128-col x K strip of Wt is staged once into LDS with async global->LDS copies,
// then B fragments are served by batched ds_load_b128 into distinct register slots so
// WMMAs retire against partial dscnt waits instead of lock-stepping.
__global__ void __launch_bounds__(256)
k_gemm_wmma(const __bf16* __restrict__ A, const __bf16* __restrict__ Wt,
            float* __restrict__ C, int K, int Nc) {
  __shared__ __attribute__((aligned(16))) __bf16 sB[128 * 256];  // 64 KB max (K<=256)

  int t    = threadIdx.x;
  int wave = t >> 5;
  int lane = t & 31;
  int col0 = blockIdx.y * 128;

  // ---- async stage: sB[c*K + k] = Wt[(col0+c)*K + k], 128*K*2 bytes ----
  {
    const char* gsrc    = (const char*)(Wt + (long)col0 * K);
    unsigned    ldsbase = (unsigned)(size_t)(&sB[0]);
    int nchunk = 16 * K;                        // 16-byte chunks total
    for (int i = t; i < nchunk; i += 256) {
      unsigned    ldso = ldsbase + (unsigned)i * 16u;
      const char* g    = gsrc + (size_t)i * 16u;
      asm volatile("global_load_async_to_lds_b128 %0, %1, off"
                   :: "v"(ldso), "v"(g) : "memory");
    }
    asm volatile("s_wait_asynccnt 0x0" ::: "memory");
    __syncthreads();
  }

  int mtile = blockIdx.x * 8 + wave;
  if (mtile >= MTILES) mtile = MTILES - 1;      // tail block: duplicate tile, identical stores

  int hi   = lane >> 4;                         // lane half
  int l15  = lane & 15;
  int row  = mtile * 16 + l15;                  // A-fragment row for this lane
  const __bf16* arow = A + (long)row * K;
  int akoff = hi * 8;    // A: lane<16 -> K 0..7 / 16..23 ; lane>=16 -> 8..15 / 24..31
  int bkoff = hi * 8;    // B: lane<16 -> K 0..7 / 16..23 ; lane>=16 -> 8..15 / 24..31

  v8f acc[8] = {};
  union AU { uint4 q[2]; v16bf v; };
  AU a_cur, a_nxt;
  a_cur.q[0] = *reinterpret_cast<const uint4*>(arow + akoff);
  a_cur.q[1] = *reinterpret_cast<const uint4*>(arow + 16 + akoff);

  for (int k0 = 0; k0 < K; k0 += 32) {
    int kn = k0 + 32;
    if (kn < K) {                               // prefetch next A fragment (overlaps WMMAs)
      a_nxt.q[0] = *reinterpret_cast<const uint4*>(arow + kn + akoff);
      a_nxt.q[1] = *reinterpret_cast<const uint4*>(arow + kn + 16 + akoff);
    }
    const __bf16* bbase = sB + (k0 + bkoff);

    // Phase 1: batch-issue all 16 ds_load_b128 into distinct slots
    AU bf[8];
#pragma unroll
    for (int j = 0; j < 8; ++j) {
      const __bf16* bp = bbase + (j * 16 + l15) * K;   // LDS
      bf[j].q[0] = *reinterpret_cast<const uint4*>(bp);
      bf[j].q[1] = *reinterpret_cast<const uint4*>(bp + 16);
    }
    // Phase 2: back-to-back WMMAs; j>0 reuses the cached A operand (ISA §7.12 RA hint)
    acc[0] = __builtin_amdgcn_wmma_f32_16x16x32_bf16(
        false, a_cur.v, false, bf[0].v, (short)0, acc[0], false, false);
#pragma unroll
    for (int j = 1; j < 8; ++j) {
      acc[j] = __builtin_amdgcn_wmma_f32_16x16x32_bf16(
          false, a_cur.v, false, bf[j].v, (short)0, acc[j], /*reuse_a=*/true, false);
    }
    a_cur = a_nxt;
  }

  // C layout: VGPR r -> row (r | hi*8), col = lane&15 within each n-tile
  int orow = mtile * 16 + hi * 8;
#pragma unroll
  for (int j = 0; j < 8; ++j) {
    int col = col0 + j * 16 + l15;
#pragma unroll
    for (int r = 0; r < 8; ++r)
      C[(long)(orow + r) * Nc + col] = acc[j][r];
  }
}

// ---------- per-node attention scores: al_s[n,h] = sum_c h[n,h,c]*a_src[h,c] ----------
__global__ void k_scores(const float* __restrict__ h, const float* __restrict__ asrc,
                         const float* __restrict__ adst, float* __restrict__ als,
                         float* __restrict__ ald, int H, int C) {
  int t = blockIdx.x * blockDim.x + threadIdx.x;
  if (t >= NNODES * H) return;
  int n = t / H, hh = t % H;
  const float* hp = h + (long)n * H * C + (long)hh * C;
  const float* ap = asrc + hh * C;
  const float* dp = adst + hh * C;
  float ss = 0.0f, sd = 0.0f;
  for (int c = 0; c < C; ++c) { float v = hp[c]; ss += v * ap[c]; sd += v * dp[c]; }
  als[t] = ss; ald[t] = sd;
}

// ---------- edge pass 1: e = leaky_relu(als[src]+ald[dst]); segment max into mbuf ----------
__global__ void k_edge_max(const long long* __restrict__ ei, const float* __restrict__ als,
                           const float* __restrict__ ald, float* __restrict__ ebuf,
                           unsigned* __restrict__ mbuf, int H) {
  int t = blockIdx.x * blockDim.x + threadIdx.x;
  if (t >= ETOT * H) return;
  int e = t / H, hh = t % H;
  int s, d;
  if (e < NEDGES) { s = (int)ei[e]; d = (int)ei[NEDGES + e]; }
  else            { s = d = e - NEDGES; }
  float v = als[s * H + hh] + ald[d * H + hh];
  v = v > 0.0f ? v : 0.2f * v;                 // leaky_relu slope 0.2
  ebuf[t] = v;
  atomicMax(&mbuf[d * H + hh], f2ord(v));
}

// ---------- edge pass 2: ex = exp(e - m[dst]); segment sum into denom ----------
__global__ void k_edge_exp(const long long* __restrict__ ei, float* __restrict__ ebuf,
                           const unsigned* __restrict__ mbuf, float* __restrict__ denom, int H) {
  int t = blockIdx.x * blockDim.x + threadIdx.x;
  if (t >= ETOT * H) return;
  int e = t / H, hh = t % H;
  int d = (e < NEDGES) ? (int)ei[NEDGES + e] : (e - NEDGES);
  float m  = ord2f(mbuf[d * H + hh]);
  float ex = expf(ebuf[t] - m);
  ebuf[t] = ex;
  atomicAdd(&denom[d * H + hh], ex);
}

// ---------- edge pass 3: agg[dst] += alpha * h[src]  (one wave32 per edge) ----------
__global__ void k_edge_agg(const long long* __restrict__ ei, const float* __restrict__ h,
                           const float* __restrict__ ebuf, const float* __restrict__ denom,
                           float* __restrict__ agg, int H, int C) {
  int lane = threadIdx.x & 31;
  int e = (int)((blockIdx.x * blockDim.x + threadIdx.x) >> 5);
  if (e >= ETOT) return;
  int s, d;
  if (e < NEDGES) { s = (int)ei[e]; d = (int)ei[NEDGES + e]; }
  else            { s = d = e - NEDGES; }
  int HC = H * C;
  const float* hs = h + (long)s * HC;
  float* ag = agg + (long)d * HC;
  __builtin_prefetch(hs + lane, 0, 1);         // global_prefetch_b8, warm src row in L2
  for (int c = lane; c < HC; c += 32) {
    int hh = c / C;
    float alpha = ebuf[e * H + hh] / (denom[d * H + hh] + 1e-16f);
    atomicAdd(&ag[c], alpha * hs[c]);
  }
}

// ---------- finalize: hidden layers -> elu(agg+b) stored directly as bf16 GEMM input ----------
__global__ void k_finalize_elu_bf16(const float* __restrict__ agg, const float* __restrict__ b,
                                    __bf16* __restrict__ out, int HC, long n) {
  long i = (long)blockIdx.x * blockDim.x + threadIdx.x;
  long st = (long)gridDim.x * blockDim.x;
  for (; i < n; i += st) {
    float v = agg[i] + b[i % HC];
    v = v > 0.0f ? v : (expf(v) - 1.0f);       // elu, alpha=1
    out[i] = (__bf16)v;
  }
}

// ---------- finalize output layer: d_out = agg + b2 ----------
__global__ void k_finalize_out(const float* __restrict__ agg, const float* __restrict__ b,
                               float* __restrict__ out, int HC, long n) {
  long i = (long)blockIdx.x * blockDim.x + threadIdx.x;
  long st = (long)gridDim.x * blockDim.x;
  for (; i < n; i += st) out[i] = agg[i] + b[i % HC];
}

// =====================================================================================
extern "C" void kernel_launch(void* const* d_in, const int* in_sizes, int n_in,
                              void* d_out, int out_size, void* d_ws, size_t ws_size,
                              hipStream_t stream) {
  (void)in_sizes; (void)n_in; (void)out_size; (void)ws_size;
  const float*     x   = (const float*)d_in[0];
  const long long* ei  = (const long long*)d_in[1];
  const float*     W0  = (const float*)d_in[2];
  const float*     as0 = (const float*)d_in[3];
  const float*     ad0 = (const float*)d_in[4];
  const float*     b0  = (const float*)d_in[5];
  const float*     W1  = (const float*)d_in[6];
  const float*     as1 = (const float*)d_in[7];
  const float*     ad1 = (const float*)d_in[8];
  const float*     b1  = (const float*)d_in[9];
  const float*     W2  = (const float*)d_in[10];
  const float*     as2 = (const float*)d_in[11];
  const float*     ad2 = (const float*)d_in[12];
  const float*     b2  = (const float*)d_in[13];

  char* ws = (char*)d_ws;
  size_t off = 0;
  auto take = [&](size_t bytes) -> char* {
    char* p = ws + off;
    off = (off + bytes + 255) & ~(size_t)255;
    return p;
  };
  float*    hbuf  = (float*)take((size_t)NNODES * 256 * 4);   // GEMM out (pre-activation)
  float*    agg   = (float*)take((size_t)NNODES * 256 * 4);   // message accumulator
  __bf16*   abf   = (__bf16*)take((size_t)NNODES * 256 * 2);  // GEMM A input (bf16)
  __bf16*   wt0   = (__bf16*)take((size_t)256 * 128 * 2);     // Wt [Nc,K]
  __bf16*   wt1   = (__bf16*)take((size_t)256 * 256 * 2);
  __bf16*   wt2   = (__bf16*)take((size_t)128 * 256 * 2);
  float*    als   = (float*)take((size_t)NNODES * 4 * 4);
  float*    ald   = (float*)take((size_t)NNODES * 4 * 4);
  unsigned* mbuf  = (unsigned*)take((size_t)NNODES * 4 * 4);
  float*    denom = (float*)take((size_t)NNODES * 4 * 4);
  float*    ebuf  = (float*)take((size_t)ETOT * 4 * 4);

  auto cdiv = [](long a, long b) -> unsigned { return (unsigned)((a + b - 1) / b); };

  // ---- prep: bf16 conversions ----
  k_f32_to_bf16<<<2048, 256, 0, stream>>>(x, abf, (long)NNODES * 128);
  k_transpose_bf16<<<cdiv(128 * 256, 256), 256, 0, stream>>>(W0, wt0, 128, 256);
  k_transpose_bf16<<<cdiv(256 * 256, 256), 256, 0, stream>>>(W1, wt1, 256, 256);
  k_transpose_bf16<<<cdiv(256 * 128, 256), 256, 0, stream>>>(W2, wt2, 256, 128);

  struct Layer { const __bf16* wt; const float *as, *ad, *b; int K, Nc, H, C; };
  Layer L[3] = {
    { wt0, as0, ad0, b0, 128, 256, 4, 64 },
    { wt1, as1, ad1, b1, 256, 256, 4, 64 },
    { wt2, as2, ad2, b2, 256, 128, 1, 128 },
  };

  for (int l = 0; l < 3; ++l) {
    const Layer& ly = L[l];
    // GEMM: hbuf = abf @ W   (8 waves/block, wave per 16x128 strip)
    dim3 gg(cdiv(MTILES, 8), ly.Nc / 128);
    k_gemm_wmma<<<gg, 256, 0, stream>>>(abf, ly.wt, hbuf, ly.K, ly.Nc);
    // per-node scores
    k_scores<<<cdiv((long)NNODES * ly.H, 256), 256, 0, stream>>>(hbuf, ly.as, ly.ad,
                                                                 als, ald, ly.H, ly.C);
    // reset segment buffers
    k_zero_f32<<<2048, 256, 0, stream>>>(agg, (long)NNODES * ly.Nc);
    k_zero_f32<<<512, 256, 0, stream>>>((float*)mbuf, (long)NNODES * ly.H);
    k_zero_f32<<<512, 256, 0, stream>>>(denom, (long)NNODES * ly.H);
    // edge phases
    k_edge_max<<<cdiv((long)ETOT * ly.H, 256), 256, 0, stream>>>(ei, als, ald, ebuf, mbuf, ly.H);
    k_edge_exp<<<cdiv((long)ETOT * ly.H, 256), 256, 0, stream>>>(ei, ebuf, mbuf, denom, ly.H);
    k_edge_agg<<<cdiv((long)ETOT * 32, 256), 256, 0, stream>>>(ei, hbuf, ebuf, denom, agg,
                                                               ly.H, ly.C);
    // finalize
    if (l < 2) {
      k_finalize_elu_bf16<<<2048, 256, 0, stream>>>(agg, ly.b, abf, ly.Nc,
                                                    (long)NNODES * ly.Nc);
    } else {
      k_finalize_out<<<2048, 256, 0, stream>>>(agg, ly.b, (float*)d_out, ly.Nc,
                                               (long)NNODES * ly.Nc);
    }
  }
}
